// ConvTemporalGraphical_5025111736621
// MI455X (gfx1250) — compile-verified, matrix operand order checked by reading, not run on hardware
//
#include <hip/hip_runtime.h>
#include <hip/hip_bf16.h>

// ---------------- problem constants ----------------
#define NB 8
#define CN 64
#define TN 12
#define VN 400
#define DD (CN * TN)   // 768
#define HEH 14
#define HNH 96

typedef __attribute__((ext_vector_type(16))) _Float16 v16h;
typedef __attribute__((ext_vector_type(8)))  _Float16 v8h;
typedef __attribute__((ext_vector_type(8)))  float    v8f;
typedef __attribute__((ext_vector_type(4)))  float    v4f;

union FragU { v16h f; v8h h[2]; };

// ---------------- edge MLP: 7 -> 14 -> 14 -> 64, masked; also A_s = mean_c ----------------
__global__ __launch_bounds__(256)
void edge_mlp_kernel(const float* __restrict__ A,
                     const float* __restrict__ We1, const float* __restrict__ be1,
                     const float* __restrict__ We2, const float* __restrict__ be2,
                     const float* __restrict__ We3, const float* __restrict__ be3,
                     float* __restrict__ A_new, float* __restrict__ A_s)
{
    __shared__ float w1[7 * HEH], b1[HEH];
    __shared__ float w2[HEH * HEH], b2[HEH];
    __shared__ float w3[HEH * CN], b3[CN];
    int tid = threadIdx.x;
    for (int i = tid; i < 7 * HEH;   i += blockDim.x) w1[i] = We1[i];
    for (int i = tid; i < HEH;       i += blockDim.x) b1[i] = be1[i];
    for (int i = tid; i < HEH * HEH; i += blockDim.x) w2[i] = We2[i];
    for (int i = tid; i < HEH;       i += blockDim.x) b2[i] = be2[i];
    for (int i = tid; i < HEH * CN;  i += blockDim.x) w3[i] = We3[i];
    for (int i = tid; i < CN;        i += blockDim.x) b3[i] = be3[i];
    __syncthreads();

    const long long VV = (long long)VN * VN;
    long long e = (long long)blockIdx.x * blockDim.x + tid;
    if (e >= (long long)NB * VV) return;
    int n = (int)(e / VV);
    long long vw = e - (long long)n * VV;

    const float* Ab = A + (long long)n * 8 * VV + vw;
    float in[7];
#pragma unroll
    for (int k = 0; k < 7; ++k) in[k] = Ab[(long long)k * VV];
    float mask = (Ab[7 * VV] > 0.5f) ? 1.0f : 0.0f;

    float h1[HEH];
#pragma unroll
    for (int j = 0; j < HEH; ++j) {
        float s = b1[j];
#pragma unroll
        for (int k = 0; k < 7; ++k) s += in[k] * w1[k * HEH + j];
        h1[j] = fmaxf(s, 0.0f);
    }
    float h2[HEH];
#pragma unroll
    for (int j = 0; j < HEH; ++j) {
        float s = b2[j];
#pragma unroll
        for (int k = 0; k < HEH; ++k) s += h1[k] * w2[k * HEH + j];
        h2[j] = fmaxf(s, 0.0f);
    }
    float acc = 0.0f;
    float* An = A_new + (long long)n * CN * VV + vw;
#pragma unroll 4
    for (int c = 0; c < CN; ++c) {
        float s = b3[c];
#pragma unroll
        for (int j = 0; j < HEH; ++j) s += h2[j] * w3[j * CN + c];
        s *= mask;
        An[(long long)c * VV] = s;
        acc += s;
    }
    A_s[(long long)n * VV + vw] = acc * (1.0f / CN);
}

// ---------------- generic batched / dual-input WMMA GEMM ----------------
// C[b] = act( A[b] @ B[b]  (+ A2 @ B2)  + bias )
// act: 0=none, 1=relu, 2=sigmoid, 3=tanh
// Wave grid WMW x WNW; each wave computes 16 rows x (NSUB*16) cols.
// Block tile: (WMW*16) x (WNW*NSUB*16).  Double-buffered LDS, 1 barrier / K-step.
#define BK 32
#define BKP 40   // padded LDS row in halves: 80B rows, 16B-aligned chunks

template <int WMW, int WNW, int NSUB>
__global__ __launch_bounds__(256)
void gemm_wmma(const float* __restrict__ A, long long sA, int lda,
               const float* __restrict__ B, long long sB, int ldb,
               const float* __restrict__ A2, int lda2,
               const float* __restrict__ B2, int ldb2, int K2,
               const float* __restrict__ bias,
               float* __restrict__ C, long long sC, int ldc,
               int M, int N, int K, int act)
{
    constexpr int BMt = WMW * 16;
    constexpr int BNt = WNW * NSUB * 16;
    constexpr int ACH = (BMt * 4 + 255) / 256;   // A row-chunks (8 elems) per thread
    constexpr int BCH = (BNt * 4 + 255) / 256;   // B col-chunks (8 elems) per thread

    __shared__ __align__(16) _Float16 As[2][BMt][BKP];
    __shared__ __align__(16) _Float16 Bs[2][BNt][BKP];

    const int tid  = threadIdx.x;
    const int lane = tid & 31;
    const int wave = tid >> 5;
    const int wm = wave / WNW;
    const int wn = wave % WNW;
    const int row0 = blockIdx.y * BMt;
    const int col0 = blockIdx.x * BNt;

    const float* Ab = A + (long long)blockIdx.z * sA;
    const float* Bb = B + (long long)blockIdx.z * sB;
    float*       Cb = C + (long long)blockIdx.z * sC;

    v8f acc[NSUB] = {};

    const int kb = (lane >> 4) * 8;        // lanes 16-31 take K+8 halves
    const int ar = wm * 16 + (lane & 15);  // fragment row in As
    const int bc0 = wn * (NSUB * 16) + (lane & 15);

    float apre[ACH][8];
    float bpre[BCH][8];

    auto kpass = [&](const float* Ap, int ldA, const float* Bp, int ldB, int Kd) {
        auto loadA = [&](int k0) {
            const bool full = (row0 + BMt <= M) && (k0 + BK <= Kd);
#pragma unroll
            for (int c = 0; c < ACH; ++c) {
                int idx = c * 256 + tid;
                if (idx < BMt * 4) {
                    int r  = idx >> 2;
                    int kc = (idx & 3) * 8;
                    int gr = row0 + r, gk = k0 + kc;
                    const float* p = Ap + (long long)gr * ldA + gk;
                    if (full) {
                        v4f x0 = *(const v4f*)p;
                        v4f x1 = *(const v4f*)(p + 4);
#pragma unroll
                        for (int j = 0; j < 4; ++j) { apre[c][j] = x0[j]; apre[c][4 + j] = x1[j]; }
                    } else {
                        bool rok = gr < M;
#pragma unroll
                        for (int j = 0; j < 8; ++j) {
                            bool ok = rok && (gk + j) < Kd;
                            long long off = ok ? ((long long)gr * ldA + gk + j) : 0;
                            float v = Ap[off];          // clamped address, no branch
                            apre[c][j] = ok ? v : 0.0f;
                        }
                    }
                }
            }
        };
        auto loadB = [&](int k0) {
            const bool full = (col0 + BNt <= N) && (k0 + BK <= Kd);
#pragma unroll
            for (int c = 0; c < BCH; ++c) {
                int idx = c * 256 + tid;
                if (idx < BNt * 4) {
                    int nn = idx % BNt;
                    int kc = (idx / BNt) * 8;
                    int gn = col0 + nn, gk0 = k0 + kc;
                    if (full) {
#pragma unroll
                        for (int j = 0; j < 8; ++j)
                            bpre[c][j] = Bp[(long long)(gk0 + j) * ldB + gn];  // coalesced across lanes
                    } else {
                        bool nok = gn < N;
#pragma unroll
                        for (int j = 0; j < 8; ++j) {
                            bool ok = nok && (gk0 + j) < Kd;
                            long long off = ok ? ((long long)(gk0 + j) * ldB + gn) : 0;
                            float v = Bp[off];
                            bpre[c][j] = ok ? v : 0.0f;
                        }
                    }
                }
            }
        };
        auto stageLDS = [&](int buf) {
#pragma unroll
            for (int c = 0; c < ACH; ++c) {
                int idx = c * 256 + tid;
                if (idx < BMt * 4) {
                    int r = idx >> 2, kc = (idx & 3) * 8;
                    v8h hv;
#pragma unroll
                    for (int j = 0; j < 8; ++j) hv[j] = (_Float16)apre[c][j];
                    *(v8h*)&As[buf][r][kc] = hv;   // one ds_store_b128
                }
            }
#pragma unroll
            for (int c = 0; c < BCH; ++c) {
                int idx = c * 256 + tid;
                if (idx < BNt * 4) {
                    int nn = idx % BNt, kc = (idx / BNt) * 8;
                    v8h hv;
#pragma unroll
                    for (int j = 0; j < 8; ++j) hv[j] = (_Float16)bpre[c][j];
                    *(v8h*)&Bs[buf][nn][kc] = hv;  // one ds_store_b128
                }
            }
        };

        // prologue: tile 0 -> buf 0
        loadA(0);
        loadB(0);
        stageLDS(0);
        __syncthreads();

        int p = 0;
        for (int k0 = 0; k0 < Kd; k0 += BK) {
            // 1) fragment loads from current buffer (WMMA waits only on DScnt)
            FragU a;
            a.h[0] = *(const v8h*)&As[p][ar][kb];
            a.h[1] = *(const v8h*)&As[p][ar][16 + kb];
            FragU b[NSUB];
#pragma unroll
            for (int t = 0; t < NSUB; ++t) {
                int col = bc0 + t * 16;
                b[t].h[0] = *(const v8h*)&Bs[p][col][kb];
                b[t].h[1] = *(const v8h*)&Bs[p][col][16 + kb];
            }
            // 2) prefetch next tile into registers (global loads fly during WMMA)
            int kn = k0 + BK;
            bool more = kn < Kd;
            if (more) { loadA(kn); loadB(kn); }
            // 3) matrix math
#pragma unroll
            for (int t = 0; t < NSUB; ++t)
                acc[t] = __builtin_amdgcn_wmma_f32_16x16x32_f16(
                    false, a.f, false, b[t].f, (short)0, acc[t], false, false);
            // 4) stage next tile into the other buffer; single barrier per step
            if (more) stageLDS(p ^ 1);
            __syncthreads();
            p ^= 1;
        }
    };

    kpass(Ab, lda, Bb, ldb, K);
    if (A2) kpass(A2, lda2, B2, ldb2, K2);

    // epilogue: D layout = 8 f32 VGPRs; lanes 0-15 -> rows r, lanes 16-31 -> rows 8+r; N = lane&15
    const int rbase = row0 + wm * 16 + (lane >> 4) * 8;
#pragma unroll
    for (int t = 0; t < NSUB; ++t) {
        int cc = col0 + bc0 + t * 16;
        float bv = (bias != nullptr && cc < N) ? bias[cc] : 0.0f;
#pragma unroll
        for (int r = 0; r < 8; ++r) {
            int rr = rbase + r;
            if (rr < M && cc < N) {
                float v = acc[t][r] + bv;
                if (act == 1)      v = fmaxf(v, 0.0f);
                else if (act == 2) v = 1.0f / (1.0f + __expf(-v));
                else if (act == 3) v = tanhf(v);
                Cb[(long long)rr * ldc + cc] = v;
            }
        }
    }
}

// ---------------- pointwise kernels ----------------
__global__ void x_to_h_kernel(const float* __restrict__ x, float* __restrict__ h, int total)
{   // h[(n*V+v)*D + d] = x[(n*D+d)*V + v]
    int idx = blockIdx.x * blockDim.x + threadIdx.x;
    if (idx >= total) return;
    int n = idx / (VN * DD);
    int r = idx - n * (VN * DD);
    int v = r / DD;
    int d = r - v * DD;
    h[idx] = x[((long long)n * DD + d) * VN + v];
}

__global__ void h_to_hf_kernel(const float* __restrict__ h, float* __restrict__ hf, int total)
{   // hf[(n*D+d)*V+v] = relu(h[(n*V+v)*D+d])
    int idx = blockIdx.x * blockDim.x + threadIdx.x;
    if (idx >= total) return;
    int n = idx / (DD * VN);
    int r = idx - n * (DD * VN);
    int d = r / VN;
    int v = r - d * VN;
    hf[idx] = fmaxf(h[((long long)n * VN + v) * DD + d], 0.0f);
}

__global__ void ew_mul_kernel(const float* __restrict__ a, const float* __restrict__ b,
                              float* __restrict__ c, int total)
{
    int i = blockIdx.x * blockDim.x + threadIdx.x;
    if (i < total) c[i] = a[i] * b[i];
}

__global__ void gru_update_kernel(float* __restrict__ h, const float* __restrict__ z,
                                  const float* __restrict__ ng, int total)
{
    int i = blockIdx.x * blockDim.x + threadIdx.x;
    if (i < total) {
        float zv = z[i];
        h[i] = (1.0f - zv) * ng[i] + zv * h[i];
    }
}

// ---------------- host driver ----------------
template <int WMW, int WNW, int NSUB>
static inline void launch_gemm(hipStream_t s,
                               const float* A, long long sA, int lda,
                               const float* B, long long sB, int ldb,
                               const float* A2, int lda2,
                               const float* B2, int ldb2, int K2,
                               const float* bias,
                               float* C, long long sC, int ldc,
                               int M, int N, int K, int act, int batches)
{
    constexpr int BMt = WMW * 16;
    constexpr int BNt = WNW * NSUB * 16;
    dim3 grid((N + BNt - 1) / BNt, (M + BMt - 1) / BMt, batches);
    gemm_wmma<WMW, WNW, NSUB><<<grid, dim3(256), 0, s>>>(A, sA, lda, B, sB, ldb,
                                                         A2, lda2, B2, ldb2, K2,
                                                         bias, C, sC, ldc, M, N, K, act);
}

extern "C" void kernel_launch(void* const* d_in, const int* in_sizes, int n_in,
                              void* d_out, int out_size, void* d_ws, size_t ws_size,
                              hipStream_t stream)
{
    (void)in_sizes; (void)n_in; (void)out_size; (void)ws_size;

    const float* x    = (const float*)d_in[0];
    const float* A    = (const float*)d_in[1];
    const float* We1  = (const float*)d_in[2];
    const float* be1  = (const float*)d_in[3];
    const float* We2  = (const float*)d_in[4];
    const float* be2  = (const float*)d_in[5];
    const float* We3  = (const float*)d_in[6];
    const float* be3  = (const float*)d_in[7];
    const float* Wn1  = (const float*)d_in[8];
    const float* bn1  = (const float*)d_in[9];
    const float* Wn2  = (const float*)d_in[10];
    const float* bn2  = (const float*)d_in[11];
    const float* Wn3  = (const float*)d_in[12];
    const float* bn3  = (const float*)d_in[13];
    const float* Wmsg = (const float*)d_in[14];
    const float* Wz   = (const float*)d_in[15];
    const float* Uz   = (const float*)d_in[16];
    const float* bz   = (const float*)d_in[17];
    const float* Wr   = (const float*)d_in[18];
    const float* Ur   = (const float*)d_in[19];
    const float* br   = (const float*)d_in[20];
    const float* Wh   = (const float*)d_in[21];
    const float* Uh   = (const float*)d_in[22];
    const float* bh   = (const float*)d_in[23];

    const long long VV   = (long long)VN * VN;           // 160000
    const long long ND   = (long long)NB * VN * DD;      // 2,457,600
    const long long OUT0 = (long long)NB * CN * TN * VN; // 2,457,600

    float* out   = (float*)d_out;          // [N,C,T,V]
    float* A_new = out + OUT0;             // [N,C,V,V] (second tuple element)

    // workspace layout (floats)
    float* ws  = (float*)d_ws;
    float* A_s = ws;                 // N*V*V
    float* h   = A_s + NB * VV;      // ND
    float* hW  = h   + ND;           // ND  (reused for r*h)
    float* m   = hW  + ND;           // ND
    float* zg  = m   + ND;           // ND
    float* rg  = zg  + ND;           // ND
    float* ng  = rg  + ND;           // ND
    float* t1  = ng  + ND;           // 3200*96
    float* t2  = t1  + 3200 * HNH;   // 3200*96
    float* hf  = t2  + 3200 * HNH;   // ND

    const int MROWS = NB * VN;       // 3200

    // 1) edge MLP -> A_new (into d_out) + A_s
    {
        long long total = (long long)NB * VV;
        int blocks = (int)((total + 255) / 256);
        edge_mlp_kernel<<<blocks, 256, 0, stream>>>(A, We1, be1, We2, be2, We3, be3,
                                                    A_new, A_s);
    }

    // 2) x -> h  ([N,V,D])
    x_to_h_kernel<<<(int)((ND + 255) / 256), 256, 0, stream>>>(x, h, (int)ND);

    // 3) node MLP: t1 = relu(h@Wn1+bn1); t2 = relu(t1@Wn2+bn2); h = t2@Wn3+bn3
    launch_gemm<4, 2, 4>(stream, h, 0, DD,  Wn1, 0, HNH, nullptr, 0, nullptr, 0, 0,
                         bn1, t1, 0, HNH, MROWS, HNH, DD, 1, 1);
    launch_gemm<4, 2, 4>(stream, t1, 0, HNH, Wn2, 0, HNH, nullptr, 0, nullptr, 0, 0,
                         bn2, t2, 0, HNH, MROWS, HNH, HNH, 1, 1);
    launch_gemm<4, 2, 4>(stream, t2, 0, HNH, Wn3, 0, DD, nullptr, 0, nullptr, 0, 0,
                         bn3, h, 0, DD, MROWS, DD, HNH, 0, 1);

    // 4) gated graph conv, 2 layers
    for (int layer = 0; layer < 2; ++layer) {
        // hW = h @ Wmsg
        launch_gemm<4, 2, 4>(stream, h, 0, DD, Wmsg, 0, DD, nullptr, 0, nullptr, 0, 0,
                             nullptr, hW, 0, DD, MROWS, DD, DD, 0, 1);
        // m[n] = A_s[n] @ hW[n]   (batched over n)
        launch_gemm<4, 2, 4>(stream, A_s, VV, VN, hW, (long long)VN * DD, DD,
                             nullptr, 0, nullptr, 0, 0,
                             nullptr, m, (long long)VN * DD, DD, VN, DD, VN, 0, NB);
        // zg = sigmoid(m@Wz + h@Uz + bz)
        launch_gemm<4, 2, 4>(stream, m, 0, DD, Wz, 0, DD, h, DD, Uz, DD, DD,
                             bz, zg, 0, DD, MROWS, DD, DD, 2, 1);
        // rg = sigmoid(m@Wr + h@Ur + br)
        launch_gemm<4, 2, 4>(stream, m, 0, DD, Wr, 0, DD, h, DD, Ur, DD, DD,
                             br, rg, 0, DD, MROWS, DD, DD, 2, 1);
        // rh = rg * h  (reuse hW)
        ew_mul_kernel<<<(int)((ND + 255) / 256), 256, 0, stream>>>(rg, h, hW, (int)ND);
        // ng = tanh(m@Wh + rh@Uh + bh)
        launch_gemm<4, 2, 4>(stream, m, 0, DD, Wh, 0, DD, hW, DD, Uh, DD, DD,
                             bh, ng, 0, DD, MROWS, DD, DD, 3, 1);
        // h = (1-zg)*ng + zg*h
        gru_update_kernel<<<(int)((ND + 255) / 256), 256, 0, stream>>>(h, zg, ng, (int)ND);
    }

    // 5) hf = relu(h) transposed to [N,D,V] ( == [N,C,T,V] )
    h_to_hf_kernel<<<(int)((ND + 255) / 256), 256, 0, stream>>>(h, hf, (int)ND);

    // 6) out[b] = hf[b] @ A_new[b], batched over b = n*C + c  (M=12 -> skinny 16x256 tile)
    launch_gemm<1, 8, 2>(stream, hf, (long long)TN * VN, VN,
                         A_new, VV, VN,
                         nullptr, 0, nullptr, 0, 0,
                         nullptr, out, (long long)TN * VN, VN,
                         TN, VN, VN, 0, NB * CN);
}
